// SparseNet_60241211294307
// MI455X (gfx1250) — matrix operand order repaired and checked by reading, not compiled, not run on hardware
//
#include <hip/hip_runtime.h>

// ---------------------------------------------------------------------------
// ISTA sparse coding on MI455X (gfx1250): bf16 WMMA (f32 acc) with f32 master
// state, TDM (tensor_load_to_lds) double-buffered tile staging.
//   E = R @ U^T - img      (GEMM1, writes bf16 E)
//   G = E @ U              (GEMM2, fused soft-threshold update of R + norms)
//   pred = R @ U^T         (final)
// Data-dependent while-loop handled via a device-side convergence flag over a
// fixed (graph-captureable) launch sequence. All reductions fixed-order.
// ---------------------------------------------------------------------------

typedef __attribute__((ext_vector_type(16))) __bf16       v16bf;
typedef __attribute__((ext_vector_type(8)))  float        v8f;
typedef __attribute__((ext_vector_type(4)))  unsigned int v4u;
typedef __attribute__((ext_vector_type(8)))  int          v8i;
typedef __attribute__((ext_vector_type(4)))  int          v4i;

#define BROWS 4096
#define NDIM  1024            // M2 == K == 1024
#define TILE  128             // block tile (M and N)
#define KSTEP 32              // WMMA K per step (bf16)
#define LDSP  40              // LDS pitch in halves: 64B data + 16B pad = 80B
#define R_LR   0.1f
#define LMDA   0.005f
#define TOL    0.01f
#define MAX_ITERS 64

#ifdef __HIP_DEVICE_COMPILE__
#  if __has_builtin(__builtin_amdgcn_tensor_load_to_lds) && \
      __has_builtin(__builtin_amdgcn_s_wait_tensorcnt)
#    define USE_TDM 1
#  endif
#endif
#ifndef USE_TDM
#  define USE_TDM 0
#endif

union F32B { uint4 u[2]; v16bf v; };

// ---------------------------------------------------------------------------
#if USE_TDM
__device__ __forceinline__ unsigned lds_off(const void* p) {
    // LDS aperture: generic addr[31:0] is the LDS byte offset.
    return (unsigned)(size_t)p;
}

// Issue one TDM load of a [128 rows x 32 halves] bf16 tile (row stride NDIM)
// into LDS with 16B padding every 64B -> LDS pitch 40 halves (bank-friendly).
__device__ __forceinline__ void tdm_tile_load(const __bf16* g, unsigned ldsoff) {
    unsigned long long ga = (unsigned long long)(size_t)g;
    v4u g0;
    g0[0] = 1u;                                                // count=1 (user D#)
    g0[1] = ldsoff;                                            // lds_addr
    g0[2] = (unsigned)ga;                                      // global_addr[31:0]
    g0[3] = (unsigned)((ga >> 32) & 0x01FFFFFFu) | (2u << 30); // addr[56:32], type=2
    v8i g1;
    g1[0] = (int)((1u << 16)   // data_size = 2 bytes
                | (1u << 20)   // pad_enable
                | (3u << 22)   // pad_interval: 16 DWORDs (64B) between pads
                | (3u << 25)); // pad_amount: 4 DWORDs (16B)
    g1[1] = (int)((unsigned)NDIM << 16);   // tensor_dim0 = 1024 (lo16<<16)
    g1[2] = (int)((unsigned)BROWS << 16);  // dim0 hi=0 | tensor_dim1 lo16<<16
    g1[3] = (int)((unsigned)KSTEP << 16);  // dim1 hi=0 | tile_dim0 = 32
    g1[4] = TILE;                          // tile_dim1 = 128, tile_dim2 = 0
    g1[5] = NDIM;                          // tensor_dim0_stride = 1024
    g1[6] = 0;                             // stride hi | dim1_stride lo
    g1[7] = 0;                             // dim1_stride hi
    v4i z4 = {0, 0, 0, 0};                 // 2D tensor: groups 2/3 unused
    v8i z8 = {0, 0, 0, 0, 0, 0, 0, 0};     // 6-arg form: trailing group zeroed
    __builtin_amdgcn_tensor_load_to_lds(g0, g1, z4, z4, z8, 0);
}
#endif

// ---- fallback cooperative global -> LDS tile copy (128 threads) ------------
__device__ __forceinline__ void stage_tile(const __bf16* __restrict__ g,
                                           int row0, int k0,
                                           __bf16* __restrict__ lds) {
    const int t = threadIdx.x;
#pragma unroll
    for (int c = 0; c < 4; ++c) {
        int idx = c * 128 + t;
        int row = idx >> 2;
        int ch  = idx & 3;
        const uint4* src = (const uint4*)(g + (size_t)(row0 + row) * NDIM + k0 + ch * 8);
        *(uint4*)(lds + row * LDSP + ch * 8) = *src;
    }
}

// ---- A fragment: 16x32 bf16. lane L: row=L&15; K={0..7,16..23} (L<16) or
// {8..15,24..31} (L>=16). ----------------------------------------------------
__device__ __forceinline__ v16bf load_a(const __bf16* __restrict__ lds,
                                        int r0, int lane) {
    int row = r0 + (lane & 15);
    int kA  = (lane >> 4) * 8;
    F32B f;
    f.u[0] = *(const uint4*)(lds + row * LDSP + kA);
    f.u[1] = *(const uint4*)(lds + row * LDSP + kA + 16);
    return f.v;
}

// ---- B fragment: 32x16 bf16. lane L: col=L&15; K={0..7,16..23} (L<16) / {8..15,24..31}. ----
__device__ __forceinline__ v16bf load_b(const __bf16* __restrict__ lds,
                                        int c0, int lane) {
    int col = c0 + (lane & 15);
    int kB  = (lane >> 4) * 8;
    F32B f;
    f.u[0] = *(const uint4*)(lds + col * LDSP + kB);
    f.u[1] = *(const uint4*)(lds + col * LDSP + kB + 16);
    return f.v;
}

// ---------------------------------------------------------------------------
// Shared GEMM main loop: acc += Ag[arow0.., :] * Bg[brow0.., :]^T over NDIM.
// sh points to 4 LDS tiles (double-buffered A/B).
// ---------------------------------------------------------------------------
__device__ __forceinline__ void gemm_tiles(const __bf16* __restrict__ Ag,
                                           const __bf16* __restrict__ Bg,
                                           int arow0, int brow0,
                                           __bf16* __restrict__ sh,
                                           v8f acc[4][4],
                                           int lane, int wr, int wc) {
    const int NS = NDIM / KSTEP;
#if USE_TDM
    const int wv = threadIdx.x >> 5;
    __bf16* t0a = sh;
    __bf16* t0b = sh + 1 * TILE * LDSP;
    __bf16* t1a = sh + 2 * TILE * LDSP;
    __bf16* t1b = sh + 3 * TILE * LDSP;
    if (wv == 0) {
        tdm_tile_load(Ag + (size_t)arow0 * NDIM, lds_off(t0a));
        tdm_tile_load(Bg + (size_t)brow0 * NDIM, lds_off(t0b));
    }
#endif
    for (int s = 0; s < NS; ++s) {
        const __bf16 *ca, *cb;
#if USE_TDM
        ca = (s & 1) ? t1a : t0a;
        cb = (s & 1) ? t1b : t0b;
        if (wv == 0) {
            if (s + 1 < NS) {      // prefetch next k-step into other buffer
                int k0 = (s + 1) * KSTEP;
                tdm_tile_load(Ag + (size_t)arow0 * NDIM + k0,
                              lds_off((s & 1) ? t0a : t1a));
                tdm_tile_load(Bg + (size_t)brow0 * NDIM + k0,
                              lds_off((s & 1) ? t0b : t1b));
                __builtin_amdgcn_s_wait_tensorcnt(2);  // current pair done
            } else {
                __builtin_amdgcn_s_wait_tensorcnt(0);
            }
        }
#else
        int k0 = s * KSTEP;
        stage_tile(Ag, arow0, k0, sh);
        stage_tile(Bg, brow0, k0, sh + TILE * LDSP);
        ca = sh;
        cb = sh + TILE * LDSP;
#endif
        __syncthreads();
        v16bf a[4], b[4];
#pragma unroll
        for (int i = 0; i < 4; ++i) a[i] = load_a(ca, wr + 16 * i, lane);
#pragma unroll
        for (int j = 0; j < 4; ++j) b[j] = load_b(cb, wc + 16 * j, lane);
#pragma unroll
        for (int i = 0; i < 4; ++i)
#pragma unroll
            for (int j = 0; j < 4; ++j)
                acc[i][j] = __builtin_amdgcn_wmma_f32_16x16x32_bf16(
                    false, a[i], false, b[j], (short)0, acc[i][j], false, false);
        __syncthreads();
    }
}

// ---------------------------------------------------------------------------
// GEMM1 / final: C = Rbf @ Ubf^T ; MODE 0: E = C - img (bf16, flag-gated)
//                                  MODE 1: pred = C (f32, always runs)
// ---------------------------------------------------------------------------
template <int MODE>
__global__ __launch_bounds__(128)
void ista_gemm_rut(const __bf16* __restrict__ Rbf,
                   const __bf16* __restrict__ Ubf,
                   const float*  __restrict__ img,
                   __bf16* __restrict__ Ebf,
                   float*  __restrict__ pred,
                   const int* __restrict__ flag) {
    if (MODE == 0) { if (*flag) return; }

    __shared__ __align__(16) __bf16 sh[4 * TILE * LDSP];   // 40 KB

    const int bbase = blockIdx.x * TILE;
    const int nbase = blockIdx.y * TILE;
    const int lane  = threadIdx.x & 31;
    const int w     = threadIdx.x >> 5;
    const int wr    = (w >> 1) * 64;
    const int wc    = (w & 1) * 64;

    v8f acc[4][4] = {};
    gemm_tiles(Rbf, Ubf, bbase, nbase, sh, acc, lane, wr, wc);

    const int rsel = (lane >> 4) * 8;   // C layout: VGPR v -> row v (+8 upper)
#pragma unroll
    for (int i = 0; i < 4; ++i)
#pragma unroll
        for (int j = 0; j < 4; ++j) {
            int col = nbase + wc + 16 * j + (lane & 15);
#pragma unroll
            for (int v = 0; v < 8; ++v) {
                int row = bbase + wr + 16 * i + rsel + v;
                size_t idx = (size_t)row * NDIM + col;
                if (MODE == 0) {
                    float e = acc[i][j][v] - img[idx];
                    Ebf[idx] = (__bf16)e;
                } else {
                    pred[idx] = acc[i][j][v];
                }
            }
        }
}

// ---------------------------------------------------------------------------
// GEMM2 + fused ISTA update: G = Ebf @ U ; Rn = shrink(R - 0.2*G, LMDA)
// ---------------------------------------------------------------------------
__global__ __launch_bounds__(128)
void ista_gemm_update(const __bf16* __restrict__ Ebf,
                      const __bf16* __restrict__ UTbf,   // [k][m] transposed
                      float*  __restrict__ Rf32,
                      __bf16* __restrict__ Rbf,
                      float*  __restrict__ partials,
                      const int* __restrict__ flag) {
    if (*flag) return;

    __shared__ __align__(16) __bf16 sh[4 * TILE * LDSP];

    const int bbase = blockIdx.x * TILE;
    const int nbase = blockIdx.y * TILE;   // output columns = codes k
    const int lane  = threadIdx.x & 31;
    const int w     = threadIdx.x >> 5;
    const int wr    = (w >> 1) * 64;
    const int wc    = (w & 1) * 64;

    v8f acc[4][4] = {};
    gemm_tiles(Ebf, UTbf, bbase, nbase, sh, acc, lane, wr, wc);

    float diffsq = 0.f, oldsq = 0.f;
    const int rsel = (lane >> 4) * 8;
#pragma unroll
    for (int i = 0; i < 4; ++i)
#pragma unroll
        for (int j = 0; j < 4; ++j) {
            int col = nbase + wc + 16 * j + (lane & 15);
#pragma unroll
            for (int v = 0; v < 8; ++v) {
                int row = bbase + wr + 16 * i + rsel + v;
                size_t idx = (size_t)row * NDIM + col;
                float g    = acc[i][j][v];
                float rold = Rf32[idx];
                float rn   = rold - 2.0f * R_LR * g;
                float y    = fmaxf(rn - LMDA, 0.f) - fmaxf(-rn - LMDA, 0.f);
                Rf32[idx] = y;
                Rbf[idx]  = (__bf16)y;
                float d = y - rold;
                diffsq += d * d;
                oldsq  += rold * rold;
            }
        }

    __syncthreads();
    float* red = (float*)sh;               // deterministic block tree reduce
    const int t = threadIdx.x;
    red[t] = diffsq;
    red[128 + t] = oldsq;
    __syncthreads();
#pragma unroll
    for (int s = 64; s > 0; s >>= 1) {
        if (t < s) { red[t] += red[t + s]; red[128 + t] += red[128 + t + s]; }
        __syncthreads();
    }
    if (t == 0) {
        int blk = blockIdx.y * gridDim.x + blockIdx.x;
        partials[2 * blk]     = red[0];
        partials[2 * blk + 1] = red[128];
    }
}

// ---- fixed-order convergence check: flag=1 once ||dR||^2 < TOL^2 ||R||^2 ---
__global__ __launch_bounds__(256)
void ista_check(const float* __restrict__ partials, int nblocks,
                int* __restrict__ flag) {
    __shared__ float d[256], o[256];
    int t = threadIdx.x;
    float dv = 0.f, ov = 0.f;
    for (int i = t; i < nblocks; i += 256) { dv += partials[2 * i]; ov += partials[2 * i + 1]; }
    d[t] = dv; o[t] = ov;
    __syncthreads();
#pragma unroll
    for (int s = 128; s > 0; s >>= 1) {
        if (t < s) { d[t] += d[t + s]; o[t] += o[t + s]; }
        __syncthreads();
    }
    if (t == 0) { if (d[0] < TOL * TOL * o[0]) *flag = 1; }
}

// ---- init: R = 0 (f32 + bf16), flag = 0 ------------------------------------
__global__ __launch_bounds__(256)
void ista_init(float* __restrict__ Rf32, __bf16* __restrict__ Rbf,
               int* __restrict__ flag) {
    size_t i = (size_t)blockIdx.x * blockDim.x + threadIdx.x;
    size_t stride = (size_t)gridDim.x * blockDim.x;
    for (size_t n = (size_t)BROWS * NDIM; i < n; i += stride) {
        Rf32[i] = 0.f;
        Rbf[i]  = (__bf16)0.f;
    }
    if (blockIdx.x == 0 && threadIdx.x == 0) *flag = 0;
}

// ---- convert U (f32, [m][k]) -> Ubf [m][k] and UTbf [k][m] ------------------
__global__ __launch_bounds__(256)
void ista_convert(const float* __restrict__ U,
                  __bf16* __restrict__ Ubf, __bf16* __restrict__ UTbf) {
    size_t idx = (size_t)blockIdx.x * blockDim.x + threadIdx.x;
    if (idx >= (size_t)NDIM * NDIM) return;
    int m = (int)(idx >> 10);
    int k = (int)(idx & (NDIM - 1));
    __bf16 v = (__bf16)U[idx];
    Ubf[idx] = v;
    UTbf[(size_t)k * NDIM + m] = v;
}

// ---------------------------------------------------------------------------
extern "C" void kernel_launch(void* const* d_in, const int* in_sizes, int n_in,
                              void* d_out, int out_size, void* d_ws, size_t ws_size,
                              hipStream_t stream) {
    const float* img = (const float*)d_in[0];   // [4096,1024]
    const float* U   = (const float*)d_in[1];   // [1024,1024]
    float* pred = (float*)d_out;                // [4096,1024]

    char* ws = (char*)d_ws;
    size_t off = 0;
    float*  Rf32 = (float*)(ws + off);  off += (size_t)BROWS * NDIM * 4;
    __bf16* Rbf  = (__bf16*)(ws + off); off += (size_t)BROWS * NDIM * 2;
    __bf16* Ebf  = (__bf16*)(ws + off); off += (size_t)BROWS * NDIM * 2;
    __bf16* Ubf  = (__bf16*)(ws + off); off += (size_t)NDIM * NDIM * 2;
    __bf16* UTbf = (__bf16*)(ws + off); off += (size_t)NDIM * NDIM * 2;
    float* partials = (float*)(ws + off); off += 2 * 256 * sizeof(float);
    int* flag = (int*)(ws + off); off += 16;

    ista_init<<<512, 256, 0, stream>>>(Rf32, Rbf, flag);
    ista_convert<<<(NDIM * NDIM) / 256, 256, 0, stream>>>(U, Ubf, UTbf);

    dim3 grid(BROWS / TILE, NDIM / TILE);   // 32 x 8 = 256 blocks
    dim3 blk(128);

    for (int it = 0; it < MAX_ITERS; ++it) {
        ista_gemm_rut<0><<<grid, blk, 0, stream>>>(Rbf, Ubf, img, Ebf, nullptr, flag);
        ista_gemm_update<<<grid, blk, 0, stream>>>(Ebf, UTbf, Rf32, Rbf, partials, flag);
        ista_check<<<1, 256, 0, stream>>>(partials, (int)(grid.x * grid.y), flag);
    }

    ista_gemm_rut<1><<<grid, blk, 0, stream>>>(Rbf, Ubf, img, nullptr, pred, flag);
}
